// AttentionPETL_8916352106743
// MI455X (gfx1250) — matrix operand order repaired and checked by reading, not compiled, not run on hardware
//
#include <hip/hip_runtime.h>
#include <hip/hip_bf16.h>
#include <hip/hip_fp16.h>

typedef __bf16 bf16_t;
typedef __attribute__((ext_vector_type(16))) __bf16 v16bf;
typedef __attribute__((ext_vector_type(8)))  __bf16 v8bf;
typedef __attribute__((ext_vector_type(8)))  float  v8f;

union V16U { v16bf v; v8bf h[2]; };

// ---------- WMMA fragment loaders (wave32, 16x16x32 bf16) ----------
// A (16x32 bf16): lane L(0-15) row M=L holds K {0..7,16..23}; lane L+16 holds K {8..15,24..31}.
__device__ __forceinline__ v16bf load_a_bf16(const bf16_t* __restrict__ base, int rowStride, int lane) {
  int r  = lane & 15;
  int kb = (lane & 16) ? 8 : 0;
  const bf16_t* p = base + (size_t)r * rowStride + kb;
  V16U u;
  u.h[0] = *reinterpret_cast<const v8bf*>(p);        // K = kb .. kb+7
  u.h[1] = *reinterpret_cast<const v8bf*>(p + 16);   // K = kb+16 .. kb+23
  return u.v;
}

// B (32x16 bf16): lane L holds column N=L%16; K = (L<16 ? {0..7,16..23} : {8..15,24..31}).
// 'base' points at B column 0 with colStride elements between columns (column-contiguous source).
__device__ __forceinline__ v16bf load_b_bf16(const bf16_t* __restrict__ base, int colStride, int lane) {
  int n  = lane & 15;
  int kb = (lane & 16) ? 8 : 0;
  const bf16_t* p = base + (size_t)n * colStride + kb;
  V16U u;
  u.h[0] = *reinterpret_cast<const v8bf*>(p);
  u.h[1] = *reinterpret_cast<const v8bf*>(p + 16);
  return u.v;
}

// A fragment from f32 source with on-the-fly bf16 convert.
__device__ __forceinline__ v16bf load_a_f32(const float* __restrict__ base, int rowStride, int lane) {
  int r  = lane & 15;
  int kb = (lane & 16) ? 8 : 0;
  const float* p = base + (size_t)r * rowStride + kb;
  v16bf a;
#pragma unroll
  for (int i = 0; i < 8; ++i) a[i] = (__bf16)p[i];
#pragma unroll
  for (int i = 0; i < 8; ++i) a[8 + i] = (__bf16)p[16 + i];
  return a;
}

__device__ __forceinline__ v8f wmma_bf16(v16bf a, v16bf b, v8f c) {
  return __builtin_amdgcn_wmma_f32_16x16x32_bf16(false, a, false, b, (short)0, c, false, false);
}

// ---------- K0: f32 -> bf16 convert ----------
__global__ __launch_bounds__(256) void k_cvt(const float* __restrict__ in,
                                             bf16_t* __restrict__ out, int n) {
  int i = blockIdx.x * 256 + threadIdx.x;
  if (i < n) out[i] = (__bf16)in[i];
}

// ---------- K1: QKV GEMM [16384x768] x [2304x768]^T + bias, scatter to q/k/vT ----------
__global__ __launch_bounds__(256) void k_qkv_gemm(
    const bf16_t* __restrict__ xb, const bf16_t* __restrict__ wb,
    const float* __restrict__ qkv_b,
    bf16_t* __restrict__ qb, bf16_t* __restrict__ kbuf, bf16_t* __restrict__ vT) {
  int wave = blockIdx.x * 8 + (threadIdx.x >> 5);
  int lane = threadIdx.x & 31;
  const int OT = 144;                       // 2304/16
  int mtile = wave / OT;
  int otile = wave - mtile * OT;
  const bf16_t* abase = xb + (size_t)mtile * 16 * 768;
  const bf16_t* bbase = wb + (size_t)otile * 16 * 768;  // B col n = qkv_w row (otile*16+n), contiguous over K
  v8f acc = {};
#pragma unroll 4
  for (int k = 0; k < 768; k += 32) {
    v16bf a = load_a_bf16(abase + k, 768, lane);
    v16bf b = load_b_bf16(bbase + k, 768, lane);
    acc = wmma_bf16(a, b, acc);
  }
  int ncol  = lane & 15;
  int obase = otile * 16;
  float bias = qkv_b[obase + ncol];
  int three = obase / 768;
  int h     = (obase % 768) / 64;
  int d     = (obase % 64) + ncol;
  int half8 = (lane >> 4) * 8;
#pragma unroll
  for (int r = 0; r < 8; ++r) {
    int m  = mtile * 16 + r + half8;
    int bi = m >> 10;
    int n  = m & 1023;
    float val = acc[r] + bias;
    if (three == 0) {
      qb[(((size_t)bi * 12 + h) * 1024 + n) * 64 + d] = (__bf16)(val * 0.125f);  // q * D^-0.5
    } else if (three == 1) {
      kbuf[(((size_t)bi * 12 + h) * 1024 + n) * 64 + d] = (__bf16)val;
    } else {
      vT[(((size_t)bi * 12 + h) * 64 + d) * 1024 + n] = (__bf16)val;             // v transposed
    }
  }
}

// ---------- K2: attn = q@k^T, blur-head mask, softmax -> f32 attn output ----------
__global__ __launch_bounds__(256) void k_attn_softmax(
    const bf16_t* __restrict__ qb, const bf16_t* __restrict__ kb,
    const int* __restrict__ blur, const int* __restrict__ target_cls,
    float* __restrict__ attn) {
  __shared__ float redmax[16][8];
  __shared__ float redsum[16][8];
  int bh    = blockIdx.x >> 6;   // b*12+h
  int rtile = blockIdx.x & 63;   // 16-row strip
  int waveId = threadIdx.x >> 5;
  int lane   = threadIdx.x & 31;
  int h  = bh % 12;
  int tc = target_cls[0];
  bool hBlur = (h == blur[0]) || (h == blur[1]);

  const bf16_t* qbase = qb + (size_t)bh * 1024 * 64 + (size_t)rtile * 16 * 64;
  v16bf a0 = load_a_bf16(qbase, 64, lane);        // K = d 0..31
  v16bf a1 = load_a_bf16(qbase + 32, 64, lane);   // K = d 32..63

  v8f accs[8];
#pragma unroll
  for (int i = 0; i < 8; ++i) {
    int j = i * 8 + waveId;                       // column tile (token block)
    const bf16_t* kbase = kb + (size_t)bh * 1024 * 64 + (size_t)j * 16 * 64;
    v16bf b0 = load_b_bf16(kbase, 64, lane);      // B col n = token row, contiguous over d
    v16bf b1 = load_b_bf16(kbase + 32, 64, lane);
    v8f c = {};
    c = wmma_bf16(a0, b0, c);
    c = wmma_bf16(a1, b1, c);
    accs[i] = c;
  }
  int half8 = (lane >> 4) * 8;

  // pre-softmax mask: zero target_cls query row for blurred heads
#pragma unroll
  for (int r = 0; r < 8; ++r) {
    int grow = rtile * 16 + r + half8;
    if (hBlur && grow == tc) {
#pragma unroll
      for (int i = 0; i < 8; ++i) accs[i][r] = 0.0f;
    }
  }

  // row max: lane-local over 8 tiles, shfl within 16-lane group, LDS across 8 waves
  float fmaxv[8];
#pragma unroll
  for (int r = 0; r < 8; ++r) {
    float m = -3.0e38f;
#pragma unroll
    for (int i = 0; i < 8; ++i) m = fmaxf(m, accs[i][r]);
    m = fmaxf(m, __shfl_xor(m, 1, 32));
    m = fmaxf(m, __shfl_xor(m, 2, 32));
    m = fmaxf(m, __shfl_xor(m, 4, 32));
    m = fmaxf(m, __shfl_xor(m, 8, 32));
    fmaxv[r] = m;
  }
  if ((lane & 15) == 0) {
#pragma unroll
    for (int r = 0; r < 8; ++r) redmax[r + half8][waveId] = fmaxv[r];
  }
  __syncthreads();
#pragma unroll
  for (int r = 0; r < 8; ++r) {
    float m = redmax[r + half8][0];
#pragma unroll
    for (int w = 1; w < 8; ++w) m = fmaxf(m, redmax[r + half8][w]);
    fmaxv[r] = m;
  }

  // exp + row sum
  float fsum[8];
#pragma unroll
  for (int r = 0; r < 8; ++r) {
    float s = 0.0f;
#pragma unroll
    for (int i = 0; i < 8; ++i) {
      float e = __expf(accs[i][r] - fmaxv[r]);
      accs[i][r] = e;
      s += e;
    }
    s += __shfl_xor(s, 1, 32);
    s += __shfl_xor(s, 2, 32);
    s += __shfl_xor(s, 4, 32);
    s += __shfl_xor(s, 8, 32);
    fsum[r] = s;
  }
  if ((lane & 15) == 0) {
#pragma unroll
    for (int r = 0; r < 8; ++r) redsum[r + half8][waveId] = fsum[r];
  }
  __syncthreads();
#pragma unroll
  for (int r = 0; r < 8; ++r) {
    float s = 0.0f;
#pragma unroll
    for (int w = 0; w < 8; ++w) s += redsum[r + half8][w];
    fsum[r] = 1.0f / s;
  }

  // normalized probabilities -> f32 attn output
  int ncol = lane & 15;
  size_t rowBase = (size_t)bh * 1024 * 1024 + (size_t)rtile * 16 * 1024;
#pragma unroll
  for (int r = 0; r < 8; ++r) {
    size_t rowOff = rowBase + (size_t)(r + half8) * 1024;
#pragma unroll
    for (int i = 0; i < 8; ++i) {
      int col = (i * 8 + waveId) * 16 + ncol;
      attn[rowOff + col] = accs[i][r] * fsum[r];
    }
  }
}

// ---------- K3: out_head = attn @ v  -> oh[B,N,C] bf16 ----------
__global__ __launch_bounds__(256) void k_attn_v(
    const float* __restrict__ attn, const bf16_t* __restrict__ vT,
    bf16_t* __restrict__ oh) {
  int wave = blockIdx.x * 8 + (threadIdx.x >> 5);
  int lane = threadIdx.x & 31;
  int bh    = wave >> 8;   // 256 tiles per (b,h)
  int t     = wave & 255;
  int mtile = t >> 2;      // 64 row tiles
  int ntile = t & 3;       // 4 col tiles (D=64)
  const float*  abase = attn + (size_t)bh * 1024 * 1024 + (size_t)mtile * 16 * 1024;
  const bf16_t* bbase = vT   + (size_t)bh * 64 * 1024  + (size_t)ntile * 16 * 1024; // B col n = d row of vT
  v8f acc = {};
  for (int k = 0; k < 1024; k += 32) {
    v16bf a = load_a_f32(abase + k, 1024, lane);
    v16bf b = load_b_bf16(bbase + k, 1024, lane);
    acc = wmma_bf16(a, b, acc);
  }
  int bi = bh / 12;
  int h  = bh % 12;
  int c  = h * 64 + ntile * 16 + (lane & 15);
  int half8 = (lane >> 4) * 8;
#pragma unroll
  for (int r = 0; r < 8; ++r) {
    int n = mtile * 16 + r + half8;
    oh[((size_t)bi * 1024 + n) * 768 + c] = (__bf16)acc[r];
  }
}

// ---------- K4: out = oh @ proj_w^T + proj_b -> f32 ----------
__global__ __launch_bounds__(256) void k_proj(
    const bf16_t* __restrict__ oh, const bf16_t* __restrict__ pw,
    const float* __restrict__ pb, float* __restrict__ out) {
  int wave = blockIdx.x * 8 + (threadIdx.x >> 5);
  int lane = threadIdx.x & 31;
  const int OT = 48;                         // 768/16
  int mtile = wave / OT;
  int otile = wave - mtile * OT;
  const bf16_t* abase = oh + (size_t)mtile * 16 * 768;
  const bf16_t* bbase = pw + (size_t)otile * 16 * 768;
  v8f acc = {};
#pragma unroll 4
  for (int k = 0; k < 768; k += 32) {
    v16bf a = load_a_bf16(abase + k, 768, lane);
    v16bf b = load_b_bf16(bbase + k, 768, lane);
    acc = wmma_bf16(a, b, acc);
  }
  int ncol = lane & 15;
  int o    = otile * 16 + ncol;
  float bias = pb[o];
  int half8 = (lane >> 4) * 8;
#pragma unroll
  for (int r = 0; r < 8; ++r) {
    int m = mtile * 16 + r + half8;
    out[(size_t)m * 768 + o] = acc[r] + bias;
  }
}

extern "C" void kernel_launch(void* const* d_in, const int* in_sizes, int n_in,
                              void* d_out, int out_size, void* d_ws, size_t ws_size,
                              hipStream_t stream) {
  const float* x      = (const float*)d_in[0];
  const float* qkv_w  = (const float*)d_in[1];
  const float* qkv_b  = (const float*)d_in[2];
  const float* proj_w = (const float*)d_in[3];
  const float* proj_b = (const float*)d_in[4];
  const int*   blur   = (const int*)d_in[5];
  const int*   tcls   = (const int*)d_in[6];
  (void)in_sizes; (void)n_in; (void)out_size; (void)ws_size;

  float* out  = (float*)d_out;
  float* attn = out + (size_t)16 * 1024 * 768;   // outputs concatenated: out then attn

  char* ws = (char*)d_ws;
  bf16_t* xb = (bf16_t*)(ws);                     // x bf16      [16384,768]   25,165,824 B
  bf16_t* wb = (bf16_t*)(ws + 25165824);          // qkv_w bf16  [2304,768]     3,538,944 B
  bf16_t* pw = (bf16_t*)(ws + 28704768);          // proj_w bf16 [768,768]      1,179,648 B
  bf16_t* qb = (bf16_t*)(ws + 29884416);          // q bf16      [B,H,N,D]     25,165,824 B
  bf16_t* kb = (bf16_t*)(ws + 55050240);          // k bf16      [B,H,N,D]     25,165,824 B
  bf16_t* vT = (bf16_t*)(ws + 80216064);          // v^T bf16    [B,H,D,N]     25,165,824 B
  bf16_t* oh = (bf16_t*)(ws + 105381888);         // out_head bf16 [B,N,C]     25,165,824 B

  k_cvt<<<(12582912 + 255) / 256, 256, 0, stream>>>(x, xb, 12582912);
  k_cvt<<<(1769472  + 255) / 256, 256, 0, stream>>>(qkv_w, wb, 1769472);
  k_cvt<<<(589824   + 255) / 256, 256, 0, stream>>>(proj_w, pw, 589824);

  k_qkv_gemm   <<<18432, 256, 0, stream>>>(xb, wb, qkv_b, qb, kb, vT);  // 1024*144 tiles
  k_attn_softmax<<<12288, 256, 0, stream>>>(qb, kb, blur, tcls, attn);  // 192*64 strips
  k_attn_v     <<<6144,  256, 0, stream>>>(attn, vT, oh);               // 192*256 tiles
  k_proj       <<<6144,  256, 0, stream>>>(oh, pw, proj_b, out);        // 1024*48 tiles
}
